// CubeMoveHead_670014898681
// MI455X (gfx1250) — matrix operand order repaired and checked by reading, not compiled, not run on hardware
//
#include <hip/hip_runtime.h>
#include <stdint.h>

// ---------------- problem constants ----------------
#define NN    262144   // num nodes
#define BB    64       // batch (graphs)
#define DD    128      // node dim
#define GG    128      // global dim
#define HH    128      // hidden
#define MM    24       // moves
#define MAXC  4096     // max cubes
#define KTOT  256      // D+G
#define NEGV  (-1e9f)
#define OUT_PER_B (MAXC * MM)   // 98304

typedef __attribute__((ext_vector_type(16))) _Float16 v16h;
typedef __attribute__((ext_vector_type(8)))  float    v8f;

// A-matrix 16x32 f16 layout (ISA 7.12.2): lane holds row M=lane%16,
// VGPR v holds K pair: k0 = (v<4?0:16) + (lane<16?0:8) + 2*(v&3)
__device__ __forceinline__ int a_k0(int lane, int v) {
    return ((v < 4) ? 0 : 16) + ((lane < 16) ? 0 : 8) + 2 * (v & 3);
}
// B-matrix 32x16 f16 layout (same K order as A): lane holds col N=lane%16,
// VGPR v holds K pair: k0 = (v<4?0:16) + (lane<16?0:8) + 2*(v&3)
__device__ __forceinline__ int b_k0(int lane, int v) {
    return ((v < 4) ? 0 : 16) + ((lane < 16) ? 0 : 8) + 2 * (v & 3);
}

// -------- pack W1 [H,256] and W2 [24,128] into WMMA B-fragment order (f16) ----
// pw1: 8 kc x 8 nt fragments; pw2: 4 kc x 2 nt fragments (cols >=24 zeroed).
// Fragment storage: ((frag*32 + lane)*16 + elem) halfs; elem 2v,2v+1 = VGPR v.
__global__ void pack_weights_kernel(const float* __restrict__ W1,
                                    const float* __restrict__ W2,
                                    _Float16* __restrict__ pw1,
                                    _Float16* __restrict__ pw2) {
    int t = blockIdx.x * blockDim.x + threadIdx.x;
    int lane = t & 31;
    int fid  = t >> 5;
    if (fid < 64) {                       // W1 fragments
        int kc = fid >> 3, nt = fid & 7;
        int n  = nt * 16 + (lane & 15);   // hidden index
        _Float16* dst = pw1 + ((size_t)fid * 32 + lane) * 16;
#pragma unroll
        for (int v = 0; v < 8; ++v) {
            int k = kc * 32 + b_k0(lane, v);
            dst[2 * v]     = (_Float16)W1[(size_t)n * KTOT + k];
            dst[2 * v + 1] = (_Float16)W1[(size_t)n * KTOT + k + 1];
        }
    } else if (fid < 72) {                // W2 fragments (padded to 32 cols)
        int f2 = fid - 64;
        int kc = f2 >> 1, nt = f2 & 1;
        int n  = nt * 16 + (lane & 15);   // move index (valid if < 24)
        _Float16* dst = pw2 + ((size_t)f2 * 32 + lane) * 16;
#pragma unroll
        for (int v = 0; v < 8; ++v) {
            int k = kc * 32 + b_k0(lane, v);
            _Float16 x0 = (_Float16)0.f, x1 = (_Float16)0.f;
            if (n < MM) {
                x0 = (_Float16)W2[(size_t)n * HH + k];
                x1 = (_Float16)W2[(size_t)n * HH + k + 1];
            }
            dst[2 * v] = x0; dst[2 * v + 1] = x1;
        }
    }
}

// -------- fill output with NEG ----------
__global__ void fill_neg_kernel(float4* __restrict__ out, int n4) {
    int i = blockIdx.x * blockDim.x + threadIdx.x;
    if (i < n4) out[i] = make_float4(NEGV, NEGV, NEGV, NEGV);
}

// -------- scan stage 1: per-block (1024 nodes) sum of cube_mask ----------
__global__ void scan_partial_kernel(const uint8_t* __restrict__ cm,
                                    int* __restrict__ partial) {
    __shared__ int sh[256];
    int t = threadIdx.x;
    int base = blockIdx.x * 1024 + t * 4;
    int s = (int)cm[base] + cm[base + 1] + cm[base + 2] + cm[base + 3];
    sh[t] = s;
    __syncthreads();
#pragma unroll
    for (int off = 128; off > 0; off >>= 1) {
        if (t < off) sh[t] += sh[t + off];
        __syncthreads();
    }
    if (t == 0) partial[blockIdx.x] = sh[0];
}

// -------- scan stage 2: single-block exclusive scan of 256 partials --------
__global__ void scan_base_kernel(int* __restrict__ partial) {
    __shared__ int sh[256];
    int t = threadIdx.x;
    sh[t] = partial[t];
    __syncthreads();
#pragma unroll
    for (int off = 1; off < 256; off <<= 1) {
        int add = (t >= off) ? sh[t - off] : 0;
        __syncthreads();
        sh[t] += add;
        __syncthreads();
    }
    partial[t] = (t > 0) ? sh[t - 1] : 0;   // exclusive block bases
}

// -------- scan stage 3: per-node exclusive cumsum + segment bases ----------
__global__ void compute_excl_kernel(const uint8_t* __restrict__ cm,
                                    const int* __restrict__ base,
                                    const int* __restrict__ batch,
                                    int* __restrict__ excl,
                                    int* __restrict__ segbase) {
    __shared__ int sh[256];
    int t = threadIdx.x;
    int i0 = blockIdx.x * 1024 + t * 4;
    int c0 = cm[i0], c1 = cm[i0 + 1], c2 = cm[i0 + 2], c3 = cm[i0 + 3];
    sh[t] = c0 + c1 + c2 + c3;
    __syncthreads();
#pragma unroll
    for (int off = 1; off < 256; off <<= 1) {
        int add = (t >= off) ? sh[t - off] : 0;
        __syncthreads();
        sh[t] += add;
        __syncthreads();
    }
    int pre = base[blockIdx.x] + ((t > 0) ? sh[t - 1] : 0);
    int e[4];
    e[0] = pre; e[1] = pre + c0; e[2] = e[1] + c1; e[3] = e[2] + c2;
    excl[i0] = e[0]; excl[i0 + 1] = e[1]; excl[i0 + 2] = e[2]; excl[i0 + 3] = e[3];
    // batch is sorted: segment base = excl at first node of each graph
#pragma unroll
    for (int j = 0; j < 4; ++j) {
        int idx = i0 + j;
        int bj = batch[idx];
        bool start = (idx == 0) || (batch[idx - 1] != bj);
        if (start) segbase[bj] = e[j];
    }
}

// -------- fused MLP (WMMA) + masked scatter ----------
// 256 threads = 8 waves; each wave owns one 16-node tile.
__global__ void __launch_bounds__(256)
fused_mlp_kernel(const float* __restrict__ nf, const float* __restrict__ gf,
                 const float* __restrict__ b1, const float* __restrict__ b2,
                 const int* __restrict__ batch, const uint8_t* __restrict__ cm,
                 const int* __restrict__ excl, const int* __restrict__ segbase,
                 const uint8_t* __restrict__ mm,
                 const _Float16* __restrict__ pw1,
                 const _Float16* __restrict__ pw2,
                 float* __restrict__ out) {
    __shared__ _Float16 hbuf[8 * 16 * HH];   // 4KB per wave: h[m][n] transpose buf

    const int lane = threadIdx.x & 31;
    const int wave = threadIdx.x >> 5;
    const int tile = blockIdx.x * 8 + wave;   // 16-node tile
    const int m0   = tile * 16;
    const int row  = lane & 15;               // A-matrix row this lane feeds
    const int node = m0 + row;
    const int gb   = batch[node];             // graph of this lane's row

    // hint the streaming row into cache early (global_prefetch_b8 path)
    __builtin_prefetch(nf + (size_t)node * DD, 0, 1);

    // ---------------- GEMM1: h = relu(x @ W1^T + b1), x = [nf | gf[batch]] ---
    v8f acc[8];
#pragma unroll
    for (int nt = 0; nt < 8; ++nt) acc[nt] = (v8f){0.f,0.f,0.f,0.f,0.f,0.f,0.f,0.f};

#pragma unroll
    for (int kc = 0; kc < 8; ++kc) {
        const float* abase = (kc < 4)
            ? (nf + (size_t)node * DD + kc * 32)
            : (gf + (size_t)gb   * GG + (kc - 4) * 32);
        v16h a;
#pragma unroll
        for (int v = 0; v < 8; ++v) {
            int k0 = a_k0(lane, v);
            float2 f = *(const float2*)(abase + k0);
            a[2 * v]     = (_Float16)f.x;
            a[2 * v + 1] = (_Float16)f.y;
        }
#pragma unroll
        for (int nt = 0; nt < 8; ++nt) {
            v16h b = *(const v16h*)(pw1 + (((size_t)kc * 8 + nt) * 32 + lane) * 16);
            acc[nt] = __builtin_amdgcn_wmma_f32_16x16x32_f16(
                false, a, false, b, (short)0, acc[nt], false, false);
        }
    }

    // C/D layout: lanes0-15 -> M=r,N=lane ; lanes16-31 -> M=8+r,N=lane-16
    const int col = lane & 15;
    const int rb  = (lane >> 4) << 3;   // 0 or 8
    _Float16* hp = hbuf + wave * (16 * HH);
#pragma unroll
    for (int nt = 0; nt < 8; ++nt) {
        float bv = b1[nt * 16 + col];
#pragma unroll
        for (int r = 0; r < 8; ++r) {
            float h = acc[nt][r] + bv;
            h = fmaxf(h, 0.f);
            hp[(rb + r) * HH + nt * 16 + col] = (_Float16)h;
        }
    }
    __syncthreads();   // C-layout -> A-layout cross-lane transpose via LDS

    // ---------------- GEMM2: scores = h @ W2^T + b2 (24 cols, padded to 32) --
    v8f acc2[2];
    acc2[0] = (v8f){0.f,0.f,0.f,0.f,0.f,0.f,0.f,0.f};
    acc2[1] = acc2[0];
#pragma unroll
    for (int kc = 0; kc < 4; ++kc) {
        const _Float16* hr = hp + row * HH + kc * 32;
        v16h a;
#pragma unroll
        for (int v = 0; v < 8; ++v) {
            int k0 = a_k0(lane, v);
            a[2 * v]     = hr[k0];
            a[2 * v + 1] = hr[k0 + 1];
        }
#pragma unroll
        for (int nt = 0; nt < 2; ++nt) {
            v16h b = *(const v16h*)(pw2 + (((size_t)kc * 2 + nt) * 32 + lane) * 16);
            acc2[nt] = __builtin_amdgcn_wmma_f32_16x16x32_f16(
                false, a, false, b, (short)0, acc2[nt], false, false);
        }
    }

    // ---------------- masked scatter into logits ----------------------------
#pragma unroll
    for (int nt = 0; nt < 2; ++nt) {
        int n2 = nt * 16 + col;
        if (n2 >= MM) continue;
        float bv = b2[n2];
#pragma unroll
        for (int r = 0; r < 8; ++r) {
            int mg = m0 + rb + r;
            if (!cm[mg]) continue;
            int bg = batch[mg];
            int rk = excl[mg] - segbase[bg];
            if (rk >= MAXC) continue;
            size_t o = (size_t)bg * OUT_PER_B + (size_t)rk * MM + n2;
            if (mm[o]) out[o] = acc2[nt][r] + bv;   // else stays NEG from fill
        }
    }
}

// ---------------- launch ----------------
extern "C" void kernel_launch(void* const* d_in, const int* in_sizes, int n_in,
                              void* d_out, int out_size, void* d_ws, size_t ws_size,
                              hipStream_t stream) {
    const float*   nf    = (const float*)d_in[0];
    const float*   gf    = (const float*)d_in[1];
    const float*   W1    = (const float*)d_in[2];
    const float*   b1    = (const float*)d_in[3];
    const float*   W2    = (const float*)d_in[4];
    const float*   b2    = (const float*)d_in[5];
    const uint8_t* cmask = (const uint8_t*)d_in[6];   // jnp bool -> 1 byte
    const int*     batch = (const int*)d_in[7];
    const uint8_t* mmask = (const uint8_t*)d_in[8];   // jnp bool -> 1 byte
    float* out = (float*)d_out;

    char* ws = (char*)d_ws;
    _Float16* pw1   = (_Float16*)(ws + 0);        // 65536 B
    _Float16* pw2   = (_Float16*)(ws + 65536);    //  8192 B
    int* partial    = (int*)(ws + 73728);         //  1024 B
    int* segb       = (int*)(ws + 74752);         //   256 B
    int* excl       = (int*)(ws + 75008);         //  1 MB

    pack_weights_kernel<<<9, 256, 0, stream>>>(W1, W2, pw1, pw2);
    fill_neg_kernel<<<6144, 256, 0, stream>>>((float4*)out, (BB * OUT_PER_B) / 4);
    scan_partial_kernel<<<256, 256, 0, stream>>>(cmask, partial);
    scan_base_kernel<<<1, 256, 0, stream>>>(partial);
    compute_excl_kernel<<<256, 256, 0, stream>>>(cmask, partial, batch, excl, segb);
    fused_mlp_kernel<<<NN / 128, 256, 0, stream>>>(nf, gf, b1, b2, batch, cmask,
                                                   excl, segb, mmask, pw1, pw2, out);
}
